// FrontierVerifierExpertHead_33784212750954
// MI455X (gfx1250) — matrix operand order repaired and checked by reading, not compiled, not run on hardware
//
#include <hip/hip_runtime.h>
#include <hip/hip_bf16.h>

// ---------------------------------------------------------------------------
// FrontierVerifierExpertHead — full forward on gfx1250 (MI455X, wave32, WMMA)
// GEMMs: 128x64x32 block tile, 8 waves x (2x2) v_wmma_f32_16x16x32_f16,
// f32->f16 staged through LDS with b128 loads/stores, fused bias/act/residual.
// ---------------------------------------------------------------------------

#define NIT   4096      // N_ITEMS
#define DIM   256       // IN_DIM
#define ODIM  10        // OUT_DIM
#define NMEM  32
#define STEPS 6
#define NSUB  5
#define NEXP  12
#define MAXE  4608
#define MAXK  5
#define ATT_SCALE 0.0625f   // 256^-0.5

#define CDIV(a,b) (((a)+(b)-1)/(b))

typedef __attribute__((ext_vector_type(16))) _Float16 v16h;
typedef __attribute__((ext_vector_type(8)))  _Float16 h8;
typedef __attribute__((ext_vector_type(8)))  float    v8f;
typedef __attribute__((ext_vector_type(4)))  float    f4;

// act ids: -1 none, 0 silu, 1 gelu(exact), 2 mish, 3 relu, 4 selu, 5 tanh,
//          6 softplus, 7 elu, 8 sigmoid
__device__ __forceinline__ float apply_act(float v, int act) {
  switch (act) {
    case 0: return v / (1.f + expf(-v));
    case 1: return 0.5f * v * (1.f + erff(v * 0.70710678118654752f));
    case 2: { float sp = (v > 0.f ? v : 0.f) + log1pf(expf(-fabsf(v)));
              return v * tanhf(sp); }
    case 3: return v > 0.f ? v : 0.f;
    case 4: { const float l = 1.0507009873554805f, a = 1.6732632423543772f;
              return v > 0.f ? l * v : l * a * expm1f(v); }
    case 5: return tanhf(v);
    case 6: return (v > 0.f ? v : 0.f) + log1pf(expf(-fabsf(v)));
    case 7: return v > 0.f ? v : expm1f(v);
    case 8: return 1.f / (1.f + expf(-v));
    default: return v;
  }
}

// ------------------------------- WMMA GEMM ---------------------------------
// C[M,N] = act(A[M,K] @ B[N,K]^T + bias) + residual.  M % 128 == 0 (M=4096).
// N,K arbitrary (edge-guarded).  f32 in/out, f16 multiply, f32 accumulate.
#define GBM 128
#define GBN 64
#define GBK 32

__global__ __launch_bounds__(256) void gemm_wmma_kernel(
    const float* __restrict__ A, const float* __restrict__ B,
    const float* __restrict__ bias, const float* __restrict__ residual,
    float* __restrict__ C, int M, int N, int K, int act)
{
  __shared__ _Float16 As[GBM][GBK];   // 8 KB
  __shared__ _Float16 Bs[GBN][GBK];   // 4 KB

  const int tid  = threadIdx.x;
  const int wave = tid >> 5;
  const int lane = tid & 31;
  const int mw   = wave & 3;    // 4 M-waves, 32 rows each
  const int nw   = wave >> 2;   // 2 N-waves, 32 cols each
  const int lm   = lane & 15;
  const int lh   = lane >> 4;

  const int bm = blockIdx.x * GBM;
  const int bn = blockIdx.y * GBN;

  v8f acc00 = {}, acc01 = {}, acc10 = {}, acc11 = {};

  for (int k0 = 0; k0 < K; k0 += GBK) {
    // ---- stage A tile (128x32): 512 chunks of 8 floats, 2 per thread ----
    #pragma unroll
    for (int it = 0; it < 2; ++it) {
      int i  = tid + it * 256;       // 0..511
      int r  = i >> 2;
      int c0 = (i & 3) * 8;
      int gk = k0 + c0;
      const float* src = A + (size_t)(bm + r) * K + gk;
      h8 hv;
      if (gk + 8 <= K) {
        f4 f0 = *(const f4*)src;
        f4 f1 = *(const f4*)(src + 4);
        hv[0] = (_Float16)f0.x; hv[1] = (_Float16)f0.y;
        hv[2] = (_Float16)f0.z; hv[3] = (_Float16)f0.w;
        hv[4] = (_Float16)f1.x; hv[5] = (_Float16)f1.y;
        hv[6] = (_Float16)f1.z; hv[7] = (_Float16)f1.w;
      } else {
        #pragma unroll
        for (int e = 0; e < 8; ++e)
          hv[e] = (gk + e < K) ? (_Float16)src[e] : (_Float16)0.f;
      }
      *(h8*)&As[r][c0] = hv;
    }
    // ---- stage B tile (64x32): 256 chunks of 8 floats, 1 per thread ----
    {
      int r  = tid >> 2;
      int c0 = (tid & 3) * 8;
      int gn = bn + r;
      int gk = k0 + c0;
      h8 hv;
      if (gn < N && gk + 8 <= K) {
        const float* src = B + (size_t)gn * K + gk;
        f4 f0 = *(const f4*)src;
        f4 f1 = *(const f4*)(src + 4);
        hv[0] = (_Float16)f0.x; hv[1] = (_Float16)f0.y;
        hv[2] = (_Float16)f0.z; hv[3] = (_Float16)f0.w;
        hv[4] = (_Float16)f1.x; hv[5] = (_Float16)f1.y;
        hv[6] = (_Float16)f1.z; hv[7] = (_Float16)f1.w;
      } else if (gn < N) {
        const float* src = B + (size_t)gn * K;
        #pragma unroll
        for (int e = 0; e < 8; ++e)
          hv[e] = (gk + e < K) ? (_Float16)src[gk + e] : (_Float16)0.f;
      } else {
        #pragma unroll
        for (int e = 0; e < 8; ++e) hv[e] = (_Float16)0.f;
      }
      *(h8*)&Bs[r][c0] = hv;
    }
    // prefetch next K-tile of A (global_prefetch_b8)
    if (k0 + GBK < K)
      __builtin_prefetch(&A[(size_t)(bm + (tid >> 1)) * K + k0 + GBK], 0, 0);
    __syncthreads();

    // ---- per-lane fragments (CDNA5 ISA layouts, CDNA5 ISA §7.12.2) ----
    v16h a0, a1, b0, b1;
    {
      const int am0 = mw * 32 + lm, am1 = am0 + 16;
      #pragma unroll
      for (int e = 0; e < 8; ++e) {
        a0[e]     = As[am0][lh * 8 + e];        // K = lh*8+e
        a0[8 + e] = As[am0][16 + lh * 8 + e];   // K = 16+lh*8+e
        a1[e]     = As[am1][lh * 8 + e];
        a1[8 + e] = As[am1][16 + lh * 8 + e];
      }
      const int bn0 = nw * 32 + lm, bn1 = bn0 + 16;
      #pragma unroll
      for (int e = 0; e < 16; ++e) {            // B: N=lm, K=lh*8+e (e<8), 16+lh*8+(e-8) (e>=8)
        b0[e] = Bs[bn0][(e < 8 ? 0 : 8) + lh * 8 + e];
        b1[e] = Bs[bn1][(e < 8 ? 0 : 8) + lh * 8 + e];
      }
    }
    acc00 = __builtin_amdgcn_wmma_f32_16x16x32_f16(false, a0, false, b0, (short)0, acc00, false, false);
    acc01 = __builtin_amdgcn_wmma_f32_16x16x32_f16(false, a0, false, b1, (short)0, acc01, false, false);
    acc10 = __builtin_amdgcn_wmma_f32_16x16x32_f16(false, a1, false, b0, (short)0, acc10, false, false);
    acc11 = __builtin_amdgcn_wmma_f32_16x16x32_f16(false, a1, false, b1, (short)0, acc11, false, false);
    __syncthreads();
  }

  // ---- epilogue: bias + act + residual, C element r: M=r+8*lh, N=lm ----
  auto store_tile = [&](const v8f& acc, int msub, int nsub) {
    int cn = bn + nw * 32 + nsub * 16 + lm;
    if (cn >= N) return;
    float bv = bias ? bias[cn] : 0.f;
    int cm0 = bm + mw * 32 + msub * 16 + 8 * lh;
    #pragma unroll
    for (int r = 0; r < 8; ++r) {
      float v = apply_act(acc[r] + bv, act);
      size_t idx = (size_t)(cm0 + r) * N + cn;
      if (residual) v += residual[idx];
      C[idx] = v;
    }
  };
  store_tile(acc00, 0, 0);
  store_tile(acc01, 0, 1);
  store_tile(acc10, 1, 0);
  store_tile(acc11, 1, 1);
}

// ----------------------------- small kernels -------------------------------
__global__ void fill_kernel(float* p, float v, int n) {
  int i = blockIdx.x * 256 + threadIdx.x;
  if (i < n) p[i] = v;
}

__global__ void add3_kernel(const float* a, const float* b, const float* c,
                            float* o, int n) {
  int i = blockIdx.x * 256 + threadIdx.x;
  if (i < n) o[i] = a[i] + b[i] + c[i];
}

__global__ __launch_bounds__(256) void ln256_kernel(
    const float* __restrict__ X, const float* __restrict__ s,
    const float* __restrict__ b, float* __restrict__ Y)
{
  __shared__ float red[256];
  const int row = blockIdx.x, t = threadIdx.x;
  float v = X[(size_t)row * 256 + t];
  red[t] = v; __syncthreads();
  for (int o = 128; o > 0; o >>= 1) { if (t < o) red[t] += red[t + o]; __syncthreads(); }
  float mu = red[0] * (1.f / 256.f); __syncthreads();
  float d = v - mu;
  red[t] = d * d; __syncthreads();
  for (int o = 128; o > 0; o >>= 1) { if (t < o) red[t] += red[t + o]; __syncthreads(); }
  float var = red[0] * (1.f / 256.f);
  Y[(size_t)row * 256 + t] = d * rsqrtf(var + 1e-5f) * s[t] + b[t];
}

__global__ void ln10_kernel(const float* X, const float* s, const float* b,
                            float* Y) {
  int n = blockIdx.x * 256 + threadIdx.x;
  if (n >= NIT) return;
  const float* x = X + (size_t)n * ODIM;
  float mu = 0.f;
  for (int j = 0; j < ODIM; ++j) mu += x[j];
  mu *= 0.1f;
  float var = 0.f;
  for (int j = 0; j < ODIM; ++j) { float d = x[j] - mu; var += d * d; }
  var *= 0.1f;
  float inv = rsqrtf(var + 1e-5f);
  for (int j = 0; j < ODIM; ++j)
    Y[(size_t)n * ODIM + j] = (x[j] - mu) * inv * s[j] + b[j];
}

__global__ void softmax_kernel(float* X, int rows, int width, float scale) {
  int r = blockIdx.x * 256 + threadIdx.x;
  if (r >= rows) return;
  float* x = X + (size_t)r * width;
  float mx = -1e30f;
  for (int j = 0; j < width; ++j) { float v = x[j] * scale; x[j] = v; mx = fmaxf(mx, v); }
  float sum = 0.f;
  for (int j = 0; j < width; ++j) { float e = expf(x[j] - mx); x[j] = e; sum += e; }
  float inv = 1.f / sum;
  for (int j = 0; j < width; ++j) x[j] *= inv;
}

// ---- rank-1 decomposed per-item memory: state = c*mem_vals + sum a_t*wv_t ---
__global__ void beta_kernel(const float* att, const float* aa, float* beta, int s) {
  int n = blockIdx.x * 256 + threadIdx.x;
  if (n >= NIT) return;
  const float* a = att + (size_t)n * NMEM;
  for (int t = 0; t < s; ++t) {
    const float* av = aa + ((size_t)t * NIT + n) * NMEM;
    float b = 0.f;
    for (int sl = 0; sl < NMEM; ++sl) b += a[sl] * av[sl];
    beta[t * NIT + n] = b;
  }
}

__global__ void memread_kernel(const float* att, const float* cc,
                               const float* memvals, const float* beta,
                               const float* wvbank, float* out, int s) {
  int i = blockIdx.x * 256 + threadIdx.x;
  if (i >= NIT * DIM) return;
  int n = i >> 8, d = i & 255;
  const float* an = att + (size_t)n * NMEM;
  const float* cn = cc + (size_t)n * NMEM;
  float acc = 0.f;
  for (int sl = 0; sl < NMEM; ++sl)
    acc += an[sl] * cn[sl] * memvals[sl * DIM + d];
  for (int t = 0; t < s; ++t)
    acc += beta[t * NIT + n] * wvbank[((size_t)t * NIT + n) * DIM + d];
  out[i] = acc;
}

__global__ void memupdate_kernel(float* cc, float* aa, const float* wg, int s) {
  int i = blockIdx.x * 256 + threadIdx.x;
  if (i >= NIT * NMEM) return;
  int n = i >> 5, sl = i & 31;
  float g = wg[i], om = 1.f - g;
  for (int t = 0; t < s; ++t) aa[((size_t)t * NIT + n) * NMEM + sl] *= om;
  aa[((size_t)s * NIT + n) * NMEM + sl] = g;
  cc[i] *= om;
}

// single-query attention over a bank of <=6 states, wave per item
__global__ __launch_bounds__(256) void attbank_kernel(
    const float* __restrict__ Q, const float* __restrict__ KB,
    const float* __restrict__ VB, float* __restrict__ OUT, int T)
{
  int wave = threadIdx.x >> 5, lane = threadIdx.x & 31;
  int n = blockIdx.x * 8 + wave;
  if (n >= NIT) return;
  const float* q = Q + (size_t)n * DIM;
  float qv[8];
  #pragma unroll
  for (int j = 0; j < 8; ++j) qv[j] = q[lane + 32 * j];
  float sc[STEPS];
  for (int t = 0; t < T; ++t) {
    const float* k = KB + ((size_t)t * NIT + n) * DIM;
    float p = 0.f;
    #pragma unroll
    for (int j = 0; j < 8; ++j) p += qv[j] * k[lane + 32 * j];
    #pragma unroll
    for (int o = 16; o > 0; o >>= 1) p += __shfl_xor(p, o, 32);
    sc[t] = p * ATT_SCALE;
  }
  float mx = -1e30f;
  for (int t = 0; t < T; ++t) mx = fmaxf(mx, sc[t]);
  float sum = 0.f;
  for (int t = 0; t < T; ++t) { sc[t] = expf(sc[t] - mx); sum += sc[t]; }
  float inv = 1.f / sum;
  float acc[8];
  #pragma unroll
  for (int j = 0; j < 8; ++j) acc[j] = 0.f;
  for (int t = 0; t < T; ++t) {
    const float* v = VB + ((size_t)t * NIT + n) * DIM;
    float w = sc[t] * inv;
    #pragma unroll
    for (int j = 0; j < 8; ++j) acc[j] += w * v[lane + 32 * j];
  }
  float* o = OUT + (size_t)n * DIM;
  #pragma unroll
  for (int j = 0; j < 8; ++j) o[lane + 32 * j] = acc[j];
}

// out[n] = (sigmoid?)(dot(A[n,:], w) + bias_arr[bias_idx]); wave per row
__global__ __launch_bounds__(256) void rowdot_kernel(
    const float* A, const float* w, const float* bias_arr, int bias_idx,
    float* out, int D, int do_sigmoid)
{
  int wave = threadIdx.x >> 5, lane = threadIdx.x & 31;
  int n = blockIdx.x * 8 + wave;
  if (n >= NIT) return;
  const float* a = A + (size_t)n * D;
  float p = 0.f;
  for (int j = lane; j < D; j += 32) p += a[j] * w[j];
  #pragma unroll
  for (int o = 16; o > 0; o >>= 1) p += __shfl_xor(p, o, 32);
  if (lane == 0) {
    float r = p + bias_arr[bias_idx];
    if (do_sigmoid) r = 1.f / (1.f + expf(-r));
    out[n] = r;
  }
}

__global__ void haltacc_kernel(const float* halt, float* cum, float* memlog,
                               float* deplog, const float* t10a,
                               const float* t10b) {
  int n = blockIdx.x * 256 + threadIdx.x;
  if (n >= NIT) return;
  float h = halt[n];
  float w = (1.f - cum[n]) * h;
  for (int j = 0; j < ODIM; ++j) {
    memlog[n * ODIM + j] += w * t10a[n * ODIM + j];
    deplog[n * ODIM + j] += w * t10b[n * ODIM + j];
  }
  cum[n] = fminf(cum[n] + h, 1.f);
}

__global__ void ssmscale_kernel(float* h, const float* decay) {
  int i = blockIdx.x * 256 + threadIdx.x;
  if (i >= NIT * 48) return;
  int j = i % 48;
  float g = 1.f / (1.f + expf(-decay[j]));
  h[i] = tanhf(h[i]) * g;
}

__global__ void concat2_kernel(const float* a, const float* b, float* o) {
  int i = blockIdx.x * 256 + threadIdx.x;
  if (i >= NIT * 512) return;
  int n = i >> 9, c = i & 511;
  o[i] = (c < 256) ? a[(size_t)n * 256 + c] : b[(size_t)n * 256 + (c - 256)];
}

__global__ void concat3_kernel(const float* a, const float* b, const float* c,
                               float* o) {
  int i = blockIdx.x * 256 + threadIdx.x;
  if (i >= NIT * 768) return;
  int n = i / 768, cc = i % 768;
  o[i] = (cc < 256) ? a[(size_t)n * 256 + cc]
       : (cc < 512) ? b[(size_t)n * 256 + (cc - 256)]
                    : c[(size_t)n * 256 + (cc - 512)];
}

__global__ void probs_kernel(const float* gate, const float* subp, float* probs) {
  int n = blockIdx.x * 256 + threadIdx.x;
  if (n >= NIT) return;
  for (int e = 0; e < NEXP; ++e) {
    float p = 0.f;
    for (int s = 0; s < NSUB; ++s)
      p += gate[n * NSUB + s] * subp[n * (NSUB * NEXP) + s * NEXP + e];
    probs[n * NEXP + e] = p;
  }
}

__global__ void budget_kernel(const float* bl, int* actk) {
  int n = blockIdx.x * 256 + threadIdx.x;
  if (n >= NIT) return;
  float best = bl[n * MAXK]; int bi = 0;
  for (int j = 1; j < MAXK; ++j)
    if (bl[n * MAXK + j] > best) { best = bl[n * MAXK + j]; bi = j; }
  actk[n] = bi + 1;
}

__global__ void topk_kernel(const float* probs, const int* actk, float* sparse) {
  int n = blockIdx.x * 256 + threadIdx.x;
  if (n >= NIT) return;
  float p[NEXP]; bool used[NEXP];
  for (int e = 0; e < NEXP; ++e) {
    p[e] = probs[n * NEXP + e]; used[e] = false; sparse[n * NEXP + e] = 0.f;
  }
  int k = actk[n];
  float total = 0.f;
  for (int r = 0; r < MAXK; ++r) {
    int bi = 0; float bv = -1e30f;
    for (int e = 0; e < NEXP; ++e)
      if (!used[e] && p[e] > bv) { bv = p[e]; bi = e; }
    used[bi] = true;
    if (r < k) { sparse[n * NEXP + bi] = bv; total += bv; }
  }
  float inv = 1.f / fmaxf(total, 1e-6f);
  for (int e = 0; e < NEXP; ++e) sparse[n * NEXP + e] *= inv;
}

__global__ void expacc_kernel(const float* sparse, const float* ln10,
                              float* expout, int i) {
  int n = blockIdx.x * 256 + threadIdx.x;
  if (n >= NIT) return;
  float w = sparse[n * NEXP + i];
  for (int j = 0; j < ODIM; ++j) expout[n * ODIM + j] += w * ln10[n * ODIM + j];
}

__global__ void final_kernel(const float* base, const float* shared_,
                             const float* aux, const float* ag,
                             const float* sscale, const float* mix,
                             const float* expout, const float* memlog,
                             const float* deplog, const float* reflog,
                             const float* rev, const float* collog,
                             const float* vg, const float* verlog,
                             const float* corrlog, float* out) {
  int n = blockIdx.x * 256 + threadIdx.x;
  if (n >= NIT) return;
  float al = tanhf(mix[0]), be = tanhf(mix[1]), ga = tanhf(mix[2]);
  float de = tanhf(mix[3]), ep = tanhf(mix[4]), ze = tanhf(mix[5]);
  float ss = sscale[0];
  float a0 = ag[n * 2], a1 = ag[n * 2 + 1];
  float v0 = vg[n * 2], v1 = vg[n * 2 + 1];
  float rv = rev[n];
  for (int j = 0; j < ODIM; ++j) {
    int id = n * ODIM + j;
    out[id] = base[id] + ss * (a0 * shared_[id] + a1 * aux[id])
            + al * expout[id] + be * memlog[id] + ze * deplog[id]
            + ga * reflog[id] + de * rv * collog[id]
            + ep * (v0 * verlog[id] + v1 * corrlog[id]);
  }
}

// ------------------------------- host side ---------------------------------
extern "C" void kernel_launch(void* const* d_in, const int* in_sizes, int n_in,
                              void* d_out, int out_size, void* d_ws,
                              size_t ws_size, hipStream_t stream) {
  const float* x             = (const float*)d_in[0];
  const float* Wm            = (const float*)d_in[1];
  const float* bvec          = (const float*)d_in[2];
  const float* shared_up_w   = (const float*)d_in[3];
  const float* shared_down_w = (const float*)d_in[4];
  const float* shared_ln_s   = (const float*)d_in[5];
  const float* shared_ln_b   = (const float*)d_in[6];
  const float* shared_scale  = (const float*)d_in[7];
  const float* mem_keys      = (const float*)d_in[8];
  const float* mem_vals      = (const float*)d_in[9];
  const float* mem_q_w       = (const float*)d_in[10];
  const float* mem_out_w     = (const float*)d_in[11];
  const float* mem_wg_w      = (const float*)d_in[12];
  const float* mem_wg_b      = (const float*)d_in[13];
  const float* mem_wv_w      = (const float*)d_in[14];
  const float* mem_wv_b      = (const float*)d_in[15];
  const float* depth_q_w     = (const float*)d_in[16];
  const float* depth_k_w     = (const float*)d_in[17];
  const float* depth_v_w     = (const float*)d_in[18];
  const float* depth_out_w   = (const float*)d_in[19];
  const float* rc_ln_s       = (const float*)d_in[20];
  const float* rc_ln_b       = (const float*)d_in[21];
  const float* rc_w1         = (const float*)d_in[22];
  const float* rc_b1         = (const float*)d_in[23];
  const float* rc_w2         = (const float*)d_in[24];
  const float* rc_b2         = (const float*)d_in[25];
  const float* ssm_in_w      = (const float*)d_in[26];
  const float* ssm_out_w     = (const float*)d_in[27];
  const float* ssm_decay     = (const float*)d_in[28];
  const float* sub_w         = (const float*)d_in[29];
  const float* sub_b         = (const float*)d_in[30];
  const float* gate_w        = (const float*)d_in[31];
  const float* gate_b        = (const float*)d_in[32];
  const float* budget_w      = (const float*)d_in[33];
  const float* budget_b      = (const float*)d_in[34];
  const float* halt_w        = (const float*)d_in[35];
  const float* halt_b        = (const float*)d_in[36];
  const float* aux_up_w      = (const float*)d_in[37];
  const float* aux_down_w    = (const float*)d_in[38];
  const float* aux_ln_s      = (const float*)d_in[39];
  const float* aux_ln_b      = (const float*)d_in[40];
  const float* aux_gate_w    = (const float*)d_in[41];
  const float* aux_gate_b    = (const float*)d_in[42];
  const float* rf_ln_s       = (const float*)d_in[43];
  const float* rf_ln_b       = (const float*)d_in[44];
  const float* rf_w1         = (const float*)d_in[45];
  const float* rf_b1         = (const float*)d_in[46];
  const float* rf_w2         = (const float*)d_in[47];
  const float* rf_b2         = (const float*)d_in[48];
  const float* reflect_out_w = (const float*)d_in[49];
  const float* coll_q_w      = (const float*)d_in[50];
  const float* coll_k_w      = (const float*)d_in[51];
  const float* coll_v_w      = (const float*)d_in[52];
  const float* coll_out_w    = (const float*)d_in[53];
  const float* revisit_w     = (const float*)d_in[54];
  const float* revisit_b     = (const float*)d_in[55];
  const float* vc_ln_s       = (const float*)d_in[56];
  const float* vc_ln_b       = (const float*)d_in[57];
  const float* vc_w1         = (const float*)d_in[58];
  const float* vc_b1         = (const float*)d_in[59];
  const float* vc_w2         = (const float*)d_in[60];
  const float* vc_b2         = (const float*)d_in[61];
  const float* ver_q_w       = (const float*)d_in[62];
  const float* ver_k_w       = (const float*)d_in[63];
  const float* ver_v_w       = (const float*)d_in[64];
  const float* ver_out_w     = (const float*)d_in[65];
  const float* ver_gate_w    = (const float*)d_in[66];
  const float* ver_gate_b    = (const float*)d_in[67];
  const float* corr_out_w    = (const float*)d_in[68];
  const float* mix           = (const float*)d_in[69];
  const float* exp_up        = (const float*)d_in[70];
  const float* exp_down      = (const float*)d_in[71];
  const float* exp_ln_s      = (const float*)d_in[72];
  const float* exp_ln_b      = (const float*)d_in[73];
  (void)in_sizes; (void)n_in; (void)out_size; (void)ws_size;

  // -------- workspace carve (floats), ~253 MB total --------
  float* w = (float*)d_ws;
  auto take = [&](size_t n) { float* p = w; w += n; return p; };
  float* CUR     = take((size_t)NIT * DIM);
  float* TMPIN   = take((size_t)NIT * DIM);
  float* TMPA    = take((size_t)NIT * DIM);
  float* TMPB    = take((size_t)NIT * DIM);
  float* TMPC    = take((size_t)NIT * DIM);
  float* MEMREAD = take((size_t)NIT * DIM);
  float* DEPTH   = take((size_t)NIT * DIM);
  float* RBUF    = take((size_t)NIT * DIM);
  float* VBUF    = take((size_t)NIT * DIM);
  float* COLLCTX = take((size_t)NIT * DIM);
  float* VERCTX  = take((size_t)NIT * DIM);
  float* CONCAT2 = take((size_t)NIT * 512);
  float* CONCAT3 = take((size_t)NIT * 768);
  float* BANK    = take((size_t)STEPS * NIT * DIM);
  float* KB      = take((size_t)STEPS * NIT * DIM);
  float* VB      = take((size_t)STEPS * NIT * DIM);
  float* WVB     = take((size_t)STEPS * NIT * DIM);
  float* HBUF    = take((size_t)NIT * MAXE);
  float* H48     = take((size_t)NIT * 48);
  float* ATT     = take((size_t)NIT * NMEM);
  float* WG      = take((size_t)NIT * NMEM);
  float* CC      = take((size_t)NIT * NMEM);
  float* AA      = take((size_t)STEPS * NIT * NMEM);
  float* BETA    = take((size_t)STEPS * NIT);
  float* ROW1    = take(NIT);
  float* CUMH    = take(NIT);
  float* REV     = take(NIT);
  float* SUB     = take((size_t)NIT * NSUB * NEXP);
  float* GATE    = take((size_t)NIT * NSUB);
  float* PROBS   = take((size_t)NIT * NEXP);
  float* SPARSE  = take((size_t)NIT * NEXP);
  float* BUD     = take((size_t)NIT * MAXK);
  float* AG      = take((size_t)NIT * 2);
  float* VG      = take((size_t)NIT * 2);
  int*   ACTK    = (int*)take(NIT);
  float* BASE    = take((size_t)NIT * ODIM);
  float* SHARED  = take((size_t)NIT * ODIM);
  float* AUX     = take((size_t)NIT * ODIM);
  float* MEMLOG  = take((size_t)NIT * ODIM);
  float* DEPLOG  = take((size_t)NIT * ODIM);
  float* EXPOUT  = take((size_t)NIT * ODIM);
  float* T10A    = take((size_t)NIT * ODIM);
  float* T10B    = take((size_t)NIT * ODIM);
  float* LN10    = take((size_t)NIT * ODIM);
  float* REFLOG  = take((size_t)NIT * ODIM);
  float* COLLOG  = take((size_t)NIT * ODIM);
  float* VERLOG  = take((size_t)NIT * ODIM);
  float* CORRLOG = take((size_t)NIT * ODIM);

  // gemm: C = act(A@B^T + bias) + residual
  auto gemm = [&](const float* A, const float* B, const float* bias,
                  const float* residual, float* C, int M, int N, int K,
                  int act) {
    dim3 g(M / GBM, CDIV(N, GBN));
    gemm_wmma_kernel<<<g, 256, 0, stream>>>(A, B, bias, residual, C, M, N, K, act);
  };
  auto fill = [&](float* p, float v, int n) {
    fill_kernel<<<CDIV(n, 256), 256, 0, stream>>>(p, v, n);
  };
  const int ND = NIT * DIM;
  // _cell(x): x + gelu(ln(x)@w1^T+b1)@w2^T + b2   (residual fused in gemm2)
  auto cell = [&](const float* Xin, float* Xout, const float* lns,
                  const float* lnb, const float* w1, const float* b1,
                  const float* w2, const float* b2, int H) {
    ln256_kernel<<<NIT, 256, 0, stream>>>(Xin, lns, lnb, TMPB);
    gemm(TMPB, w1, b1, nullptr, HBUF, NIT, H, DIM, 1 /*gelu*/);
    gemm(HBUF, w2, b2, Xin /*residual*/, Xout, NIT, DIM, H, -1);
  };

  // ---------------- init ----------------
  hipMemcpyAsync(CUR, x, sizeof(float) * ND, hipMemcpyDeviceToDevice, stream);
  fill(CC, 1.0f, NIT * NMEM);
  fill(CUMH, 0.f, NIT);
  fill(MEMLOG, 0.f, NIT * ODIM);
  fill(DEPLOG, 0.f, NIT * ODIM);
  fill(EXPOUT, 0.f, NIT * ODIM);

  // ---------------- x-only branches ----------------
  gemm(x, Wm, bvec, nullptr, BASE, NIT, ODIM, DIM, -1);              // base
  gemm(x, shared_up_w, nullptr, nullptr, HBUF, NIT, 3072, DIM, 0);   // silu fused
  gemm(HBUF, shared_down_w, nullptr, nullptr, T10A, NIT, ODIM, 3072, -1);
  ln10_kernel<<<CDIV(NIT, 256), 256, 0, stream>>>(T10A, shared_ln_s, shared_ln_b, SHARED);
  gemm(x, aux_up_w, nullptr, nullptr, HBUF, NIT, 4096, DIM, 1);      // gelu fused
  gemm(HBUF, aux_down_w, nullptr, nullptr, T10A, NIT, ODIM, 4096, -1);
  ln10_kernel<<<CDIV(NIT, 256), 256, 0, stream>>>(T10A, aux_ln_s, aux_ln_b, AUX);
  gemm(x, aux_gate_w, aux_gate_b, nullptr, AG, NIT, 2, DIM, -1);     // ag
  softmax_kernel<<<CDIV(NIT, 256), 256, 0, stream>>>(AG, NIT, 2, 1.f);

  // ---------------- recurrent loop ----------------
  for (int s = 0; s < STEPS; ++s) {
    // memory read (rank-1 decomposed state)
    gemm(CUR, mem_q_w, nullptr, nullptr, TMPA, NIT, DIM, DIM, -1);   // q
    gemm(TMPA, mem_keys, nullptr, nullptr, ATT, NIT, NMEM, DIM, -1); // q@K^T
    softmax_kernel<<<CDIV(NIT, 256), 256, 0, stream>>>(ATT, NIT, NMEM, ATT_SCALE);
    if (s > 0)
      beta_kernel<<<CDIV(NIT, 256), 256, 0, stream>>>(ATT, AA, BETA, s);
    memread_kernel<<<CDIV(ND, 256), 256, 0, stream>>>(ATT, CC, mem_vals, BETA,
                                                      WVB, MEMREAD, s);
    // gated write (sigmoid fused into gemm epilogue)
    concat2_kernel<<<CDIV(NIT * 512, 256), 256, 0, stream>>>(CUR, MEMREAD, CONCAT2);
    gemm(CONCAT2, mem_wg_w, mem_wg_b, nullptr, WG, NIT, NMEM, 512, 8);
    gemm(CUR, mem_wv_w, mem_wv_b, nullptr, WVB + (size_t)s * ND, NIT, DIM, DIM, -1);
    memupdate_kernel<<<CDIV(NIT * NMEM, 256), 256, 0, stream>>>(CC, AA, WG, s);
    // depth attention over previous states
    if (s == 0) {
      fill(DEPTH, 0.f, ND);
    } else {
      gemm(CUR, depth_q_w, nullptr, nullptr, TMPA, NIT, DIM, DIM, -1);
      attbank_kernel<<<NIT / 8, 256, 0, stream>>>(TMPA, KB, VB, DEPTH, s);
    }
    // cell + ssm (residuals fused)
    add3_kernel<<<CDIV(ND, 256), 256, 0, stream>>>(CUR, MEMREAD, DEPTH, TMPIN, ND);
    cell(TMPIN, CUR, rc_ln_s + (size_t)s * DIM, rc_ln_b + (size_t)s * DIM,
         rc_w1 + (size_t)s * 896 * DIM, rc_b1 + (size_t)s * 896,
         rc_w2 + (size_t)s * DIM * 896, rc_b2 + (size_t)s * DIM, 896);
    gemm(CUR, ssm_in_w, nullptr, nullptr, H48, NIT, 48, DIM, -1);
    ssmscale_kernel<<<CDIV(NIT * 48, 256), 256, 0, stream>>>(H48, ssm_decay);
    gemm(H48, ssm_out_w, nullptr, CUR /*residual*/, CUR, NIT, DIM, 48, -1);
    // append to bank + precompute depth K/V of the new state
    hipMemcpyAsync(BANK + (size_t)s * ND, CUR, sizeof(float) * ND,
                   hipMemcpyDeviceToDevice, stream);
    gemm(CUR, depth_k_w, nullptr, nullptr, KB + (size_t)s * ND, NIT, DIM, DIM, -1);
    gemm(CUR, depth_v_w, nullptr, nullptr, VB + (size_t)s * ND, NIT, DIM, DIM, -1);
    // halting-weighted logit accumulation
    rowdot_kernel<<<NIT / 8, 256, 0, stream>>>(CUR, halt_w + (size_t)s * DIM,
                                               halt_b, s, ROW1, DIM, 1);
    gemm(MEMREAD, mem_out_w, nullptr, nullptr, T10A, NIT, ODIM, DIM, -1);
    gemm(DEPTH, depth_out_w, nullptr, nullptr, T10B, NIT, ODIM, DIM, -1);
    haltacc_kernel<<<CDIV(NIT, 256), 256, 0, stream>>>(ROW1, CUMH, MEMLOG,
                                                       DEPLOG, T10A, T10B);
  }

  // ---------------- routing ----------------
  gemm(CUR, sub_w, sub_b, nullptr, SUB, NIT, NSUB * NEXP, DIM, -1);
  softmax_kernel<<<CDIV(NIT * NSUB, 256), 256, 0, stream>>>(SUB, NIT * NSUB, NEXP, 1.f);
  gemm(CUR, gate_w, gate_b, nullptr, GATE, NIT, NSUB, DIM, -1);
  softmax_kernel<<<CDIV(NIT, 256), 256, 0, stream>>>(GATE, NIT, NSUB, 1.f);
  probs_kernel<<<CDIV(NIT, 256), 256, 0, stream>>>(GATE, SUB, PROBS);
  gemm(CUR, budget_w, budget_b, nullptr, BUD, NIT, MAXK, DIM, -1);
  budget_kernel<<<CDIV(NIT, 256), 256, 0, stream>>>(BUD, ACTK);
  topk_kernel<<<CDIV(NIT, 256), 256, 0, stream>>>(PROBS, ACTK, SPARSE);

  // ---------------- experts (FLOP hot spot, act fused into WMMA GEMM) ------
  for (int i = 0; i < NEXP; ++i) {
    gemm(CUR, exp_up + (size_t)i * MAXE * DIM, nullptr, nullptr, HBUF,
         NIT, MAXE, DIM, i % 8);
    gemm(HBUF, exp_down + (size_t)i * ODIM * MAXE, nullptr, nullptr, T10A,
         NIT, ODIM, MAXE, -1);
    ln10_kernel<<<CDIV(NIT, 256), 256, 0, stream>>>(
        T10A, exp_ln_s + (size_t)i * ODIM, exp_ln_b + (size_t)i * ODIM, LN10);
    expacc_kernel<<<CDIV(NIT, 256), 256, 0, stream>>>(SPARSE, LN10, EXPOUT, i);
  }

  // ---------------- reflection ----------------
  hipMemcpyAsync(RBUF, CUR, sizeof(float) * ND, hipMemcpyDeviceToDevice, stream);
  for (int i = 0; i < 3; ++i)
    cell(RBUF, RBUF, rf_ln_s + (size_t)i * DIM, rf_ln_b + (size_t)i * DIM,
         rf_w1 + (size_t)i * 1280 * DIM, rf_b1 + (size_t)i * 1280,
         rf_w2 + (size_t)i * DIM * 1280, rf_b2 + (size_t)i * DIM, 1280);
  gemm(RBUF, reflect_out_w, nullptr, nullptr, REFLOG, NIT, ODIM, DIM, -1);

  // ---------------- collective attention + revisit ----------------
  for (int t = 0; t < STEPS; ++t) {           // reuse KB/VB for coll K/V
    gemm(BANK + (size_t)t * ND, coll_k_w, nullptr, nullptr, KB + (size_t)t * ND, NIT, DIM, DIM, -1);
    gemm(BANK + (size_t)t * ND, coll_v_w, nullptr, nullptr, VB + (size_t)t * ND, NIT, DIM, DIM, -1);
  }
  gemm(CUR, coll_q_w, nullptr, nullptr, TMPA, NIT, DIM, DIM, -1);
  attbank_kernel<<<NIT / 8, 256, 0, stream>>>(TMPA, KB, VB, COLLCTX, STEPS);
  gemm(COLLCTX, coll_out_w, nullptr, nullptr, COLLOG, NIT, ODIM, DIM, -1);
  concat3_kernel<<<CDIV(NIT * 768, 256), 256, 0, stream>>>(x, CUR, COLLCTX, CONCAT3);
  rowdot_kernel<<<NIT / 8, 256, 0, stream>>>(CONCAT3, revisit_w, revisit_b, 0,
                                             REV, 768, 1);

  // ---------------- verifier + correction ----------------
  hipMemcpyAsync(VBUF, CUR, sizeof(float) * ND, hipMemcpyDeviceToDevice, stream);
  for (int i = 0; i < 2; ++i)
    cell(VBUF, VBUF, vc_ln_s + (size_t)i * DIM, vc_ln_b + (size_t)i * DIM,
         vc_w1 + (size_t)i * 1536 * DIM, vc_b1 + (size_t)i * 1536,
         vc_w2 + (size_t)i * DIM * 1536, vc_b2 + (size_t)i * DIM, 1536);
  for (int t = 0; t < STEPS; ++t) {           // reuse KB/VB for ver K/V
    gemm(BANK + (size_t)t * ND, ver_k_w, nullptr, nullptr, KB + (size_t)t * ND, NIT, DIM, DIM, -1);
    gemm(BANK + (size_t)t * ND, ver_v_w, nullptr, nullptr, VB + (size_t)t * ND, NIT, DIM, DIM, -1);
  }
  gemm(VBUF, ver_q_w, nullptr, nullptr, TMPA, NIT, DIM, DIM, -1);
  attbank_kernel<<<NIT / 8, 256, 0, stream>>>(TMPA, KB, VB, VERCTX, STEPS);
  concat2_kernel<<<CDIV(NIT * 512, 256), 256, 0, stream>>>(VBUF, VERCTX, CONCAT2);
  gemm(CONCAT2, ver_gate_w, ver_gate_b, nullptr, VG, NIT, 2, 512, -1);
  softmax_kernel<<<CDIV(NIT, 256), 256, 0, stream>>>(VG, NIT, 2, 1.f);
  gemm(VERCTX, ver_out_w, nullptr, nullptr, VERLOG, NIT, ODIM, DIM, -1);
  gemm(VBUF, corr_out_w, nullptr, nullptr, CORRLOG, NIT, ODIM, DIM, -1);

  // ---------------- final mix ----------------
  final_kernel<<<CDIV(NIT, 256), 256, 0, stream>>>(
      BASE, SHARED, AUX, AG, shared_scale, mix, EXPOUT, MEMLOG, DEPLOG,
      REFLOG, REV, COLLOG, VG, VERLOG, CORRLOG, (float*)d_out);
}
